// DynamicConvCaps_29970281791752
// MI455X (gfx1250) — matrix operand order repaired and checked by reading, not compiled, not run on hardware
//
#include <hip/hip_runtime.h>

typedef __attribute__((ext_vector_type(16))) _Float16 v16h;
typedef __attribute__((ext_vector_type(4)))  _Float16 v4h;
typedef __attribute__((ext_vector_type(8)))  float    v8f;

// Problem constants
#define NB    2
#define HH    14
#define WW    14
#define CIN   544      // B*(PSIZE+1) = 32*17
#define BCAP  32       // B
#define NI    288      // K*K*B
#define NC    32       // C
#define PS    16       // PSIZE
#define NPIX  392      // NB*HH*WW
#define NPAD  400      // padded to 25 tiles of 16
#define NT    25
#define EPSR  1e-6f
#define NWELEM (NI * NC * PS * PS)                  // 2359296 weight elems

// d_out layout (float32): [p_out 392*512][a_out 392*32][out 392*544]
#define P_OFF   0
#define A_OFF   (NPIX * NC * PS)                    // 200704
#define O_OFF   (A_OFF + NPIX * NC)                 // 213248

// d_ws layout (bytes)
#define PIN_BYTES   ((size_t)NPAD * NI * PS * 2)    // fp16 p_in   = 3,686,400
#define AIN_BYTES   ((size_t)NPAD * NI * 4)         // fp32 a_in   =   460,800
#define WH_BYTES    ((size_t)NWELEM * 2)            // fp16 W      = 4,718,592
#define WH_OFF      (PIN_BYTES + AIN_BYTES)
#define VOT_OFF     (WH_OFF + WH_BYTES)
#define VROW_ELEMS  (NI * NC * PS)                  // 147456 f16 per position

// LDS layout for routing kernel (bytes)
#define SH_VOTES  0                                  // 288*32*16 f16 = 294912
#define SH_CIJ    294912                             // 288*32 f16    = 18432
#define SH_A      313344                             // 288 f32       = 1152
#define SH_S      314496                             // 512 f32       = 2048
#define SH_V      316544                             // 512 f32       = 2048
#define SH_S2     318592                             // 32 f32        = 128
#define SH_BYTES  318720                             // < 320KB WGP LDS

typedef __attribute__((address_space(3))) char lds_char;

// ---------------------------------------------------------------------------
// Kernel 0: weights f32 -> f16 (done once; removes 25x redundant converts)
// ---------------------------------------------------------------------------
__global__ __launch_bounds__(256)
void k_wcvt(const float* __restrict__ w, _Float16* __restrict__ wh) {
  int t = blockIdx.x * blockDim.x + threadIdx.x;
  if (t >= NWELEM / 4) return;
  float4 v = ((const float4*)w)[t];
  v4h h;
  h[0] = (_Float16)v.x; h[1] = (_Float16)v.y;
  h[2] = (_Float16)v.z; h[3] = (_Float16)v.w;
  ((v4h*)wh)[t] = h;
}

// ---------------------------------------------------------------------------
// Kernel 1: scrambled unfold (replicates F.unfold + channel-major view)
// p_in[n][i][p] (fp16), a_in[n][i] (fp32); rows n>=392 zeroed.
// ---------------------------------------------------------------------------
__global__ __launch_bounds__(256)
void k_unfold(const float* __restrict__ x,
              _Float16* __restrict__ p_in,
              float* __restrict__ a_in) {
  int t = blockIdx.x * blockDim.x + threadIdx.x;
  if (t >= NPAD * NI) return;
  int n = t / NI, i = t % NI;
  if (n >= NPIX) {
    #pragma unroll
    for (int p = 0; p < PS; ++p) p_in[(size_t)t * PS + p] = (_Float16)0.f;
    a_in[t] = 0.f;
    return;
  }
  int kk = i / BCAP, cap = i % BCAP;
  int b = n / (HH * WW), rem = n % (HH * WW);
  int y = rem / WW, xc = rem % WW;
  #pragma unroll
  for (int p = 0; p < PS; ++p) {
    int flat = kk * CIN + cap * PS + p;       // channel-major reinterpretation
    int cidx = flat / 9, r = flat % 9;
    int ki = r / 3, kj = r % 3;
    int yy = y + ki - 1, xx = xc + kj - 1;
    float v = 0.f;
    if (yy >= 0 && yy < HH && xx >= 0 && xx < WW)
      v = x[(((size_t)b * HH + yy) * WW + xx) * CIN + cidx];
    p_in[(size_t)t * PS + p] = (_Float16)v;
  }
  {
    int flat = kk * CIN + BCAP * PS + cap;    // activation channel
    int cidx = flat / 9, r = flat % 9;
    int ki = r / 3, kj = r % 3;
    int yy = y + ki - 1, xx = xc + kj - 1;
    float v = 0.f;
    if (yy >= 0 && yy < HH && xx >= 0 && xx < WW)
      v = x[(((size_t)b * HH + yy) * WW + xx) * CIN + cidx];
    a_in[t] = v;
  }
}

// ---------------------------------------------------------------------------
// Kernel 2: votes via v_wmma_f32_16x16x32_f16.
// Block = (n-tile nt, vote i); 128 threads = 4 waves; wave w handles 8 c's.
// W[i] tile (16KB f16) staged in LDS; fragments built with unconditional
// LDS reads + value selects (no exec divergence); stores unpredicated
// (votes padded to NPAD rows).
// ---------------------------------------------------------------------------
__global__ __launch_bounds__(128)
void k_votes(const _Float16* __restrict__ p_in,
             const _Float16* __restrict__ wh,
             _Float16* __restrict__ votes) {
  __shared__ _Float16 P[16][16];                // 512 B
  __shared__ _Float16 Wsh[NC * PS * PS];        // 16 KB
  int nt = blockIdx.x, i = blockIdx.y;
  int tid = threadIdx.x;

  for (int e = tid; e < 256; e += 128) {
    int m = e >> 4, p = e & 15;
    P[m][p] = p_in[((size_t)(nt * 16 + m) * NI + i) * PS + p];
  }
  {
    const uint4* src = (const uint4*)(wh + (size_t)i * NC * PS * PS);
    uint4* dst = (uint4*)Wsh;
    #pragma unroll
    for (int k = 0; k < 8; ++k) dst[tid + 128 * k] = src[tid + 128 * k];
  }
  __syncthreads();

  int lane = tid & 31, wv = tid >> 5;
  bool hi = lane >= 16;
  int mrow = lane & 15;

  // A fragment: contiguous K order: lanes 0-15 hold K 0..15 in slots 0..15;
  // lanes 16-31 hold K 16..31 (zero).
  v16h A;
  #pragma unroll
  for (int j = 0; j < 16; ++j) {
    int K = j + (hi ? 16 : 0);
    _Float16 pv = P[mrow][K & 15];              // unconditional LDS read
    A[j] = (K < PS) ? pv : (_Float16)0.f;       // v_cndmask, no branch
  }

  _Float16* vout = votes + ((size_t)(nt * 16) * NI + i) * NC * PS;
  for (int cc = 0; cc < 8; ++cc) {
    int c = wv * 8 + cc;
    // B fragment: lanes 0-15 hold K=0..15 (N=lane); lanes 16-31 K>=16 -> 0.
    v16h Bf;
    #pragma unroll
    for (int j = 0; j < 16; ++j) {
      _Float16 wvv = Wsh[c * 256 + j * PS + mrow];  // unconditional LDS read
      Bf[j] = hi ? (_Float16)0.f : wvv;
    }
    v8f acc = {};
    acc = __builtin_amdgcn_wmma_f32_16x16x32_f16(false, A, false, Bf,
                                                 (short)0, acc, false, false);
    // D layout: VGPR r -> M = r (+8 for lanes 16-31), N = lane%16 = q
    #pragma unroll
    for (int r = 0; r < 8; ++r) {
      int m = r + (hi ? 8 : 0);
      vout[(size_t)m * (NI * NC * PS) + (size_t)c * PS + mrow] = (_Float16)acc[r];
    }
  }
}

// ---------------------------------------------------------------------------
// Kernel 3: dynamic routing, one block per spatial position, 256 threads.
// Votes (288KB fp16) DMA'd into LDS via gfx1250 async-to-LDS ops and kept
// resident for all 3 iterations (needs CDNA5 320KB WGP LDS).
// b_ij in registers: thread t owns row t, and row 256+t if t<32.
// ---------------------------------------------------------------------------
__device__ __forceinline__
void compute_cij(const float* b, float a, _Float16* crow) {
  float m = b[0];
  #pragma unroll
  for (int c = 1; c < NC; ++c) m = fmaxf(m, b[c]);
  float s = 0.f;
  #pragma unroll
  for (int c = 0; c < NC; ++c) s += __expf(b[c] - m);
  float scale = a / (s * (a + EPSR));   // softmax*a then /(sum_c + eps)
  #pragma unroll
  for (int c = 0; c < NC; ++c) crow[c] = (_Float16)(__expf(b[c] - m) * scale);
}

__global__ __launch_bounds__(256)
void k_route(const _Float16* __restrict__ votes,
             const float* __restrict__ a_in,
             float* __restrict__ out) {
  extern __shared__ char smem[];
  _Float16* vsh = (_Float16*)(smem + SH_VOTES);
  _Float16* csh = (_Float16*)(smem + SH_CIJ);
  float*    ash = (float*)(smem + SH_A);
  float*    ssh = (float*)(smem + SH_S);
  float*    vjs = (float*)(smem + SH_V);
  float*    s2s = (float*)(smem + SH_S2);

  int n = blockIdx.x, tid = threadIdx.x;
  const _Float16* vrow = votes + (size_t)n * VROW_ELEMS;

  // Stage votes (288KB) into LDS with CDNA5 async DMA (no VGPR round-trip).
  // global_load_async_to_lds_b128: per-lane 16B global->LDS, ASYNCcnt-tracked.
  {
    unsigned ldsbase = (unsigned)(unsigned long long)(lds_char*)vsh;
    const char* gbase = (const char*)vrow;
    for (int idx = tid; idx < 18432; idx += 256) {
      unsigned la = ldsbase + (unsigned)idx * 16u;
      unsigned long long ga = (unsigned long long)(gbase + (size_t)idx * 16);
      asm volatile("global_load_async_to_lds_b128 %0, %1, off"
                   :: "v"(la), "v"(ga) : "memory");
    }
  }
  for (int i = tid; i < NI; i += 256) {
    float a = a_in[(size_t)n * NI + i];
    ash[i] = fminf(fmaxf(a, 1e-4f), 1.0f);   // clip(a_in, 1e-4, 1)
  }
  asm volatile("s_wait_asynccnt 0x0" ::: "memory");
  __syncthreads();

  float b0[NC], b1[NC];
  #pragma unroll
  for (int c = 0; c < NC; ++c) { b0[c] = 0.f; b1[c] = 0.f; }
  int row0 = tid;
  int row1 = 256 + tid;           // only valid for tid < 32

  for (int it = 0; it < 3; ++it) {
    compute_cij(b0, ash[row0], &csh[row0 * NC]);
    if (tid < 32) compute_cij(b1, ash[row1], &csh[row1 * NC]);
    ssh[tid] = 0.f; ssh[tid + 256] = 0.f;
    __syncthreads();

    // s_j[c][q] = sum_i c_ij[i][c] * votes[i][c][q]; 2 entries per thread
    #pragma unroll
    for (int k = 0; k < 2; ++k) {
      int e = tid * 2 + k;
      int c = e >> 4, q = e & 15;
      float acc = 0.f;
      for (int i2 = 0; i2 < NI; ++i2)
        acc += (float)csh[i2 * NC + c] * (float)vsh[((size_t)i2 * NC + c) * PS + q];
      ssh[e] = acc;
    }
    __syncthreads();

    if (tid < NC) {
      float s2 = 0.f;
      #pragma unroll
      for (int q = 0; q < PS; ++q) { float v = ssh[tid * PS + q]; s2 += v * v; }
      s2s[tid] = s2;
    }
    __syncthreads();

    // squash
    #pragma unroll
    for (int k = 0; k < 2; ++k) {
      int e = tid * 2 + k;
      float s2 = s2s[e >> 4];
      float scale = (s2 / (1.f + s2)) * rsqrtf(s2 + EPSR);
      vjs[e] = scale * ssh[e];
    }
    __syncthreads();

    if (it < 2) {   // agreement: b_ij += <votes[i,c,:], v_j[c,:]>
      #pragma unroll 4
      for (int c = 0; c < NC; ++c) {
        float d = 0.f;
        #pragma unroll
        for (int q = 0; q < PS; ++q)
          d += (float)vsh[((size_t)row0 * NC + c) * PS + q] * vjs[c * PS + q];
        b0[c] += d;
      }
      if (tid < 32) {
        #pragma unroll 4
        for (int c = 0; c < NC; ++c) {
          float d = 0.f;
          #pragma unroll
          for (int q = 0; q < PS; ++q)
            d += (float)vsh[((size_t)row1 * NC + c) * PS + q] * vjs[c * PS + q];
          b1[c] += d;
        }
      }
      __syncthreads();
    }
  }

  // Outputs: p_out, a_out, and concatenated out, with nan_to_num semantics.
  #pragma unroll
  for (int k = 0; k < 2; ++k) {
    int e = tid * 2 + k;
    float v = vjs[e];
    if (__builtin_isnan(v)) v = 0.f;
    else if (__builtin_isinf(v)) v = (v > 0.f) ? 10000.f : -10000.f;
    out[P_OFF + (size_t)n * (NC * PS) + e] = v;
    out[O_OFF + (size_t)n * CIN + e] = v;
  }
  if (tid < NC) {
    float s2 = 0.f;
    #pragma unroll
    for (int q = 0; q < PS; ++q) { float v = vjs[tid * PS + q]; s2 += v * v; }
    float a = sqrtf(s2 + EPSR);
    if (__builtin_isnan(a)) a = 0.5f;
    a = fminf(fmaxf(a, 1e-4f), 1.f - 1e-4f);
    out[A_OFF + (size_t)n * NC + tid] = a;
    out[O_OFF + (size_t)n * CIN + NC * PS + tid] = a;
  }
}

// ---------------------------------------------------------------------------
extern "C" void kernel_launch(void* const* d_in, const int* in_sizes, int n_in,
                              void* d_out, int out_size, void* d_ws, size_t ws_size,
                              hipStream_t stream) {
  const float* x = (const float*)d_in[0];        // (2,14,14,544) f32
  const float* w = (const float*)d_in[1];        // (288,32,16,16) f32
  float* out = (float*)d_out;                    // 426496 f32

  _Float16* p_in  = (_Float16*)d_ws;
  float*    a_in  = (float*)((char*)d_ws + PIN_BYTES);
  _Float16* wh    = (_Float16*)((char*)d_ws + WH_OFF);
  _Float16* votes = (_Float16*)((char*)d_ws + VOT_OFF);

  (void)in_sizes; (void)n_in; (void)out_size; (void)ws_size;

  // Allow 318KB dynamic LDS (CDNA5 WGP has 320KB).
  (void)hipFuncSetAttribute((const void*)k_route,
                            hipFuncAttributeMaxDynamicSharedMemorySize,
                            SH_BYTES);

  k_wcvt<<<(NWELEM / 4 + 255) / 256, 256, 0, stream>>>(w, wh);
  k_unfold<<<(NPAD * NI + 255) / 256, 256, 0, stream>>>(x, p_in, a_in);
  k_votes<<<dim3(NT, NI), 128, 0, stream>>>(p_in, wh, votes);
  k_route<<<NPIX, 256, SH_BYTES, stream>>>(votes, a_in, out);
}
